// DiffusionConvLayer_12240656793874
// MI455X (gfx1250) — matrix operand order repaired and checked by reading, not compiled, not run on hardware
//
#include <hip/hip_runtime.h>
#include <hip/hip_bf16.h>
#include <stdint.h>

#define NN 1024   // nodes
#define TT 64     // time steps
#define BB 16     // batch
#define CC 32     // channels in
#define GG (BB*CC)
#define COUT 32
#define CIN_TOT 224

#define KC 128          // K chunk staged in LDS
#define LDS_STRIDE 136  // bf16 elems per LDS row (128 + 8 pad -> conflict-free)

typedef __attribute__((ext_vector_type(16))) __bf16 v16bf;
typedef __attribute__((ext_vector_type(8)))  __bf16 v8bf;
typedef __attribute__((ext_vector_type(8)))  float  v8f;
typedef __attribute__((ext_vector_type(4)))  int    v4i;

#if defined(__has_builtin)
#  if __has_builtin(__builtin_amdgcn_global_load_async_to_lds_b128)
#    define HAVE_ASYNC_LDS 1
#  endif
#endif
#ifndef HAVE_ASYNC_LDS
#  define HAVE_ASYNC_LDS 0
#endif

__device__ __forceinline__ unsigned short f32_to_bf16(float f) {
  union { float f; unsigned int u; } x; x.f = f;
  unsigned int u = x.u;
  u += 0x7fffu + ((u >> 16) & 1u);
  return (unsigned short)(u >> 16);
}
__device__ __forceinline__ float bf16_to_f32(unsigned short h) {
  union { unsigned int u; float f; } x; x.u = ((unsigned int)h) << 16; return x.f;
}

#if HAVE_ASYNC_LDS
// 16B global -> LDS async copy (TENSORless async path, tracked by ASYNCcnt).
// Param 1 per compiler diagnostic: v4i addrspace(1)*; param 2: LDS address.
// Build AS-qualified pointers via integer casts (flat->AS1 is identity;
// flat LDS low 32 bits are the AS3 offset per the aperture rules).
typedef __attribute__((address_space(1))) v4i* gas_v4i_p;
typedef __attribute__((address_space(3))) v4i* las_v4i_p;
__device__ __forceinline__ void async_copy_b128(const void* g, void* lds) {
  gas_v4i_p gp = (gas_v4i_p)(unsigned long long)(uintptr_t)g;
  las_v4i_p lp = (las_v4i_p)(unsigned int)(uintptr_t)lds;
  __builtin_amdgcn_global_load_async_to_lds_b128(gp, lp, 0, 0);
}
__device__ __forceinline__ void wait_async0() {
#if defined(__has_builtin) && __has_builtin(__builtin_amdgcn_s_wait_asynccnt)
  __builtin_amdgcn_s_wait_asynccnt(0);
#else
  asm volatile("s_wait_asynccnt 0x0" ::: "memory");
#endif
}
#endif

// ---------------------------------------------------------------------------
// adaptive adjacency: adp = softmax(mask(relu(nv1 @ nv2)), axis=1)   [N,N] f32
// ---------------------------------------------------------------------------
__global__ void k_adp(const float* __restrict__ nv1, const float* __restrict__ nv2,
                      float* __restrict__ adp) {
  __shared__ float red[256];
  const int row = blockIdx.x;
  const int tid = threadIdx.x;
  float nv1r[10];
#pragma unroll
  for (int d = 0; d < 10; ++d) nv1r[d] = nv1[row * 10 + d];
  float v[4];
#pragma unroll
  for (int i = 0; i < 4; ++i) {
    int col = tid + i * 256;
    float s = 0.f;
#pragma unroll
    for (int d = 0; d < 10; ++d) s += nv1r[d] * nv2[d * NN + col];
    v[i] = (s > 0.f) ? s : -1e10f;   // relu + mask exact zeros
  }
  float m = fmaxf(fmaxf(v[0], v[1]), fmaxf(v[2], v[3]));
  red[tid] = m; __syncthreads();
  for (int off = 128; off > 0; off >>= 1) {
    if (tid < off) red[tid] = fmaxf(red[tid], red[tid + off]);
    __syncthreads();
  }
  m = red[0]; __syncthreads();
  float e[4]; float s = 0.f;
#pragma unroll
  for (int i = 0; i < 4; ++i) { e[i] = __expf(v[i] - m); s += e[i]; }
  red[tid] = s; __syncthreads();
  for (int off = 128; off > 0; off >>= 1) {
    if (tid < off) red[tid] += red[tid + off];
    __syncthreads();
  }
  float inv = 1.f / red[0];
#pragma unroll
  for (int i = 0; i < 4; ++i) adp[row * NN + tid + i * 256] = e[i] * inv;
}

// ---------------------------------------------------------------------------
// AT[w,v] = a[v,w], fp32 -> bf16, tiled via LDS
// ---------------------------------------------------------------------------
__global__ void k_at(const float* __restrict__ in, unsigned short* __restrict__ out) {
  __shared__ float tile[32][33];
  const int bx = blockIdx.x * 32;
  const int by = blockIdx.y * 32;
  const int tx = threadIdx.x;
  const int ty = threadIdx.y;
#pragma unroll
  for (int i = 0; i < 32; i += 8)
    tile[ty + i][tx] = in[(size_t)(bx + ty + i) * NN + by + tx];
  __syncthreads();
#pragma unroll
  for (int i = 0; i < 32; i += 8)
    out[(size_t)(by + ty + i) * NN + bx + tx] = f32_to_bf16(tile[tx][ty + i]);
}

// ---------------------------------------------------------------------------
// x [G, N, T] f32  ->  xb [G, T, N] bf16   (per-g N x T transpose, LDS tiled)
// ---------------------------------------------------------------------------
__global__ void k_xlt(const float* __restrict__ x, unsigned short* __restrict__ xb) {
  __shared__ float tile[64][65];
  const int g  = blockIdx.y;
  const int v0 = blockIdx.x * 64;
  const float* xg = x + (size_t)g * NN * TT;
  unsigned short* xo = xb + (size_t)g * TT * NN;
  const int tid = threadIdx.x;
#pragma unroll
  for (int k = 0; k < 16; ++k) {
    int idx = tid + k * 256;
    int vl = idx >> 6, ll = idx & 63;
    tile[vl][ll] = xg[(size_t)(v0 + vl) * TT + ll];
  }
  __syncthreads();
#pragma unroll
  for (int k = 0; k < 16; ++k) {
    int idx = tid + k * 256;
    int ll = idx >> 6, vl = idx & 63;
    xo[(size_t)ll * NN + v0 + vl] = f32_to_bf16(tile[vl][ll]);
  }
}

// ---------------------------------------------------------------------------
// WMMA hop helpers
// ---------------------------------------------------------------------------
__device__ __forceinline__ v16bf load_afrag(const __bf16* p) {
  // A k order used here (contiguous, k = 16h + s, same as B): this lane's K range is p[0..15]
  v8bf lo = *(const v8bf*)(p);
  v8bf hi = *(const v8bf*)(p + 8);
  v16bf a;
#pragma unroll
  for (int i = 0; i < 8; ++i) { a[i] = lo[i]; a[i + 8] = hi[i]; }
  return a;
}
__device__ __forceinline__ v16bf load_bfrag_lds(const __bf16* p) {
  v8bf lo = *(const v8bf*)(p);
  v8bf hi = *(const v8bf*)(p + 8);
  v16bf b;
#pragma unroll
  for (int i = 0; i < 8; ++i) { b[i] = lo[i]; b[i + 8] = hi[i]; }
  return b;
}
__device__ __forceinline__ void store_tile(v8f c, unsigned short* Zg, int rb, int col) {
#pragma unroll
  for (int r = 0; r < 8; ++r)
    Zg[(size_t)(rb + r) * NN + col] = f32_to_bf16(c[r]);
}

// ---------------------------------------------------------------------------
// One diffusion hop in LT layout:  Z[g,l,w] = sum_v Y[g,l,v] * AT[w,v]
// 8 waves/block; each wave owns a 32x32 macro-tile (2x2 WMMA tiles -> 4 accs).
// B (AT) chunks staged in LDS via async-to-LDS copies.
// Block covers l = 0..63 (all of T), w = blockIdx.x*128 .. +128.
// ---------------------------------------------------------------------------
__global__ void __launch_bounds__(256) k_hop(const unsigned short* __restrict__ Yp,
                                             const unsigned short* __restrict__ ATp,
                                             unsigned short* __restrict__ Zp) {
  __shared__ __bf16 Bs[128 * LDS_STRIDE];   // ~34.8 KB

  const int tid  = threadIdx.x;
  const int lane = tid & 31;
  const int wave = tid >> 5;
  const int g     = blockIdx.z;
  const int wbase = blockIdx.x * 128;
  const int l0    = (wave >> 2) * 32;   // 0 or 32
  const int w0loc = (wave & 3) * 32;    // 0,32,64,96 within block

  const __bf16* Yg = (const __bf16*)Yp + (size_t)g * TT * NN;
  const __bf16* AT = (const __bf16*)ATp;

  const int akoff = (lane < 16) ? 0 : 16;
  const __bf16* Abase0 = Yg + (size_t)(l0 + (lane & 15)) * NN + akoff;
  const __bf16* Abase1 = Abase0 + (size_t)16 * NN;

  const int bkoff = (lane < 16) ? 0 : 16;
  const __bf16* Bs0 = Bs + (size_t)(w0loc + (lane & 15)) * LDS_STRIDE + bkoff;
  const __bf16* Bs1 = Bs0 + (size_t)16 * LDS_STRIDE;

  // staging map: thread copies rows (tid>>4)+16i at column offset (tid&15)*8
  const int row0 = tid >> 4;
  const int coff = (tid & 15) * 8;

  v8f c00 = {}, c01 = {}, c10 = {}, c11 = {};

  for (int kc = 0; kc < NN; kc += KC) {
    __syncthreads();   // previous chunk's readers done before overwrite
#if HAVE_ASYNC_LDS
#pragma unroll
    for (int i = 0; i < 8; ++i) {
      const int r = row0 + 16 * i;
      async_copy_b128(AT + (size_t)(wbase + r) * NN + kc + coff,
                      &Bs[r * LDS_STRIDE + coff]);
    }
    wait_async0();
#else
#pragma unroll
    for (int i = 0; i < 8; ++i) {
      const int r = row0 + 16 * i;
      *(v8bf*)(&Bs[r * LDS_STRIDE + coff]) =
          *(const v8bf*)(AT + (size_t)(wbase + r) * NN + kc + coff);
    }
#endif
    __builtin_prefetch(Abase0 + kc + KC, 0, 0);   // next A chunk
    __syncthreads();

#pragma unroll
    for (int k0 = 0; k0 < KC; k0 += 32) {
      v16bf a0 = load_afrag(Abase0 + kc + k0);
      v16bf a1 = load_afrag(Abase1 + kc + k0);
      v16bf b0 = load_bfrag_lds(Bs0 + k0);
      v16bf b1 = load_bfrag_lds(Bs1 + k0);
      c00 = __builtin_amdgcn_wmma_f32_16x16x32_bf16(false, a0, false, b0, (short)0, c00, false, false);
      c01 = __builtin_amdgcn_wmma_f32_16x16x32_bf16(false, a0, false, b1, (short)0, c01, false, false);
      c10 = __builtin_amdgcn_wmma_f32_16x16x32_bf16(false, a1, false, b0, (short)0, c10, false, false);
      c11 = __builtin_amdgcn_wmma_f32_16x16x32_bf16(false, a1, false, b1, (short)0, c11, false, false);
    }
  }

  unsigned short* Zg = Zp + (size_t)g * TT * NN;
  const int colb = wbase + w0loc + (lane & 15);
  const int rb   = l0 + ((lane < 16) ? 0 : 8);
  store_tile(c00, Zg, rb,      colb);
  store_tile(c01, Zg, rb,      colb + 16);
  store_tile(c10, Zg, rb + 16, colb);
  store_tile(c11, Zg, rb + 16, colb + 16);
}

// ---------------------------------------------------------------------------
// out[b,o,v,l] (+)= sum_c W[o, jblock*32 + c] * H[(b,c),l,v]   (+ bias if init)
// ---------------------------------------------------------------------------
__global__ void __launch_bounds__(256) k_mix(const unsigned short* __restrict__ H,
                                             const float* __restrict__ W,
                                             const float* __restrict__ bias,
                                             float* __restrict__ out,
                                             int jblock, int init) {
  __shared__ float Wl[COUT * CC];
  __shared__ float bl[COUT];
  const int tid = threadIdx.x;
  for (int i = tid; i < COUT * CC; i += 256) {
    int o = i >> 5, cc = i & 31;
    Wl[i] = W[o * CIN_TOT + jblock * CC + cc];
  }
  if (tid < COUT) bl[tid] = init ? bias[tid] : 0.f;
  __syncthreads();

  const size_t id = (size_t)blockIdx.x * 256 + tid;   // over B*T*N = 1M
  const int v = (int)(id & (NN - 1));
  const int l = (int)((id >> 10) & (TT - 1));
  const int b = (int)(id >> 16);

  float acc[COUT];
#pragma unroll
  for (int o = 0; o < COUT; ++o) acc[o] = 0.f;

  const unsigned short* Hb = H + ((size_t)(b * CC) * TT + l) * NN + v;
#pragma unroll 4
  for (int cc = 0; cc < CC; ++cc) {
    float hv = bf16_to_f32(Hb[(size_t)cc * TT * NN]);
#pragma unroll
    for (int o = 0; o < COUT; ++o) acc[o] += Wl[o * CC + cc] * hv;
  }

  float* ob = out + (((size_t)b * COUT) * NN + v) * TT + l;  // o stride = NN*TT
  if (init) {
#pragma unroll
    for (int o = 0; o < COUT; ++o) ob[(size_t)o * NN * TT] = acc[o] + bl[o];
  } else {
#pragma unroll
    for (int o = 0; o < COUT; ++o) ob[(size_t)o * NN * TT] += acc[o];
  }
}

// ---------------------------------------------------------------------------
extern "C" void kernel_launch(void* const* d_in, const int* in_sizes, int n_in,
                              void* d_out, int out_size, void* d_ws, size_t ws_size,
                              hipStream_t stream) {
  const float* x   = (const float*)d_in[0];
  const float* A1  = (const float*)d_in[1];
  const float* A2  = (const float*)d_in[2];
  const float* nv1 = (const float*)d_in[3];
  const float* nv2 = (const float*)d_in[4];
  const float* W   = (const float*)d_in[5];
  const float* b   = (const float*)d_in[6];
  float* out = (float*)d_out;

  char* ws = (char*)d_ws;
  const size_t MB = 1024ull * 1024ull;
  float*          adp  = (float*)(ws + 0);               //  4 MB
  unsigned short* AT1  = (unsigned short*)(ws + 4*MB);   //  2 MB
  unsigned short* AT2  = (unsigned short*)(ws + 6*MB);   //  2 MB
  unsigned short* ATad = (unsigned short*)(ws + 8*MB);   //  2 MB
  unsigned short* xb   = (unsigned short*)(ws + 16*MB);  // 64 MB
  unsigned short* buf1 = (unsigned short*)(ws + 80*MB);  // 64 MB
  unsigned short* buf2 = (unsigned short*)(ws + 144*MB); // 64 MB (total 208 MB)

  k_adp<<<NN, 256, 0, stream>>>(nv1, nv2, adp);
  k_at<<<dim3(32, 32), dim3(32, 8), 0, stream>>>(A1,  AT1);
  k_at<<<dim3(32, 32), dim3(32, 8), 0, stream>>>(A2,  AT2);
  k_at<<<dim3(32, 32), dim3(32, 8), 0, stream>>>(adp, ATad);

  k_xlt<<<dim3(NN / 64, GG), 256, 0, stream>>>(x, xb);

  const int mixBlocks = (BB * TT * NN) / 256;  // 4096
  const dim3 hopGrid(NN / 128, 1, GG);         // (8,1,512), 8 waves/block

  k_mix<<<mixBlocks, 256, 0, stream>>>(xb, W, b, out, 0, 1);

  k_hop<<<hopGrid, 256, 0, stream>>>(xb,   AT1, buf1);
  k_mix<<<mixBlocks, 256, 0, stream>>>(buf1, W, b, out, 1, 0);
  k_hop<<<hopGrid, 256, 0, stream>>>(buf1, AT1, buf2);
  k_mix<<<mixBlocks, 256, 0, stream>>>(buf2, W, b, out, 2, 0);

  k_hop<<<hopGrid, 256, 0, stream>>>(xb,   AT2, buf1);
  k_mix<<<mixBlocks, 256, 0, stream>>>(buf1, W, b, out, 3, 0);
  k_hop<<<hopGrid, 256, 0, stream>>>(buf1, AT2, buf2);
  k_mix<<<mixBlocks, 256, 0, stream>>>(buf2, W, b, out, 4, 0);

  k_hop<<<hopGrid, 256, 0, stream>>>(xb,   ATad, buf1);
  k_mix<<<mixBlocks, 256, 0, stream>>>(buf1, W, b, out, 5, 0);
  k_hop<<<hopGrid, 256, 0, stream>>>(buf1, ATad, buf2);
  k_mix<<<mixBlocks, 256, 0, stream>>>(buf2, W, b, out, 6, 0);
}
